// SelfAttention_78778290143895
// MI455X (gfx1250) — matrix operand order repaired and checked by reading, not compiled, not run on hardware
//
#include <hip/hip_runtime.h>
#include <hip/hip_bf16.h>

typedef __attribute__((ext_vector_type(16))) __bf16         v16bf;
typedef __attribute__((ext_vector_type(8)))  float          v8f;
typedef __attribute__((ext_vector_type(4)))  unsigned int   u32x4;

#define BATCH   8
#define NTOK    4096
#define MTOT    (BATCH * NTOK)      // 32768 total rows
#define CDIM    256
#define DHEAD   32
#define KCHUNK  32
#define QR_WAVE 16
#define WAVES   4
#define QR_BLK  (QR_WAVE * WAVES)

union Frag {               // 16 bf16 == 8 dwords == 2 x 16B
  v16bf bf;
  u32x4 q[2];
};

__device__ __forceinline__ unsigned short f2bf(float f) {
  unsigned int u = __builtin_bit_cast(unsigned int, f);
  u += 0x7fffu + ((u >> 16) & 1u);          // round-to-nearest-even
  return (unsigned short)(u >> 16);
}

// 16 contiguous bf16 -> fragment (B-frags, and 32-wide rows)
__device__ __forceinline__ v16bf load_frag32(const unsigned short* p) {
  Frag f;
  f.q[0] = *(const u32x4*)(p);
  f.q[1] = *(const u32x4*)(p + 8);
  return f.bf;
}
// A-frag: 16 contiguous k at row+2*hi8 (k = 16*half + s, matches B-frags)
__device__ __forceinline__ v16bf load_frag_a(const unsigned short* row, int hi8) {
  Frag f;
  f.q[0] = *(const u32x4*)(row + 2 * hi8);
  f.q[1] = *(const u32x4*)(row + 2 * hi8 + 8);
  return f.bf;
}

// ---------------------------------------------------------------- stage 1
// x: fp32 -> bf16, 2 elements per thread, packed dword stores.
__global__ void cvt_f32_bf16x2(const float* __restrict__ src,
                               unsigned int* __restrict__ dst, int npairs) {
  int i = blockIdx.x * blockDim.x + threadIdx.x;
  if (i < npairs) {
    float a = src[2 * i], b = src[2 * i + 1];
    dst[i] = (unsigned int)f2bf(a) | ((unsigned int)f2bf(b) << 16);
  }
}

// W[C x OC] fp32 -> Wt[OC x C] bf16 (transposed so B-frags are contiguous)
__global__ void cvt_w_transpose(const float* __restrict__ W,
                                unsigned short* __restrict__ Wt,
                                int C, int OC) {
  int i = blockIdx.x * blockDim.x + threadIdx.x;
  if (i < C * OC) {
    int c = i / OC, oc = i % OC;
    Wt[(size_t)oc * C + c] = f2bf(W[i]);
  }
}

// ---------------------------------------------------------------- stage 2
// Out = A[M x 256] * Wt[OC x 256]^T (bf16 in, fp32 accum, bf16 out).
// One wave per 16x16 output tile. out_tr==0: Out[row*OC+col];
// out_tr!=0: Out[col*MTOT+row]  (pre-transposed for the flash V stream).
__global__ __launch_bounds__(128)
void gemm_bf16(const unsigned short* __restrict__ A,
               const unsigned short* __restrict__ Wt,   // [OC x 256]
               unsigned short* __restrict__ Out,
               int OC, int tiles_n, int out_tr) {
  int wid  = (blockIdx.x * blockDim.x + threadIdx.x) >> 5;
  int lane = threadIdx.x & 31;
  int tm = wid / tiles_n, tn = wid % tiles_n;
  int half = lane >> 4, l15 = lane & 15, hi8 = half * 8;

  const unsigned short* arow = A + (size_t)(tm * 16 + l15) * CDIM;
  const unsigned short* wrow = Wt + (size_t)(tn * 16 + l15) * CDIM + half * 16;
  v8f acc = {0.f, 0.f, 0.f, 0.f, 0.f, 0.f, 0.f, 0.f};

#pragma unroll
  for (int kt = 0; kt < CDIM / 32; ++kt) {
    v16bf a = load_frag_a(arow + kt * 32, hi8);      // 2 x global_load_b128
    v16bf b = load_frag32(wrow + kt * 32);           // 2 x global_load_b128
    acc = __builtin_amdgcn_wmma_f32_16x16x32_bf16(
        false, a, false, b, (short)0, acc, false, false);
  }
#pragma unroll
  for (int r = 0; r < 8; ++r) {                      // C/D layout: row r+8*half
    int row = tm * 16 + r + hi8;
    int col = tn * 16 + l15;
    size_t idx = out_tr ? ((size_t)col * MTOT + row) : ((size_t)row * OC + col);
    Out[idx] = f2bf(acc[r]);
  }
}

// ---------------------------------------------------------------- stage 3+4
// Flash attention + fused residual epilogue.
__global__ __launch_bounds__(128)
void flash_attn(const unsigned short* __restrict__ qb,   // [B*N x 32]
                const unsigned short* __restrict__ kb,   // [B*N x 32]
                const unsigned short* __restrict__ vT,   // [256 x B*N] pre-transposed
                const float* __restrict__ x,             // [B*N x 256]
                const float* __restrict__ gamma,
                float* __restrict__ out) {
  __shared__ __attribute__((aligned(16))) unsigned short Klds[KCHUNK * DHEAD];  // [key][d]
  __shared__ __attribute__((aligned(16))) unsigned short Vt[CDIM * KCHUNK];     // [ch][key]
  __shared__ __attribute__((aligned(16))) unsigned short Plds[WAVES][16 * KCHUNK];

  const int b = blockIdx.y;
  const int tid = threadIdx.x;
  const int wave = tid >> 5, lane = tid & 31;
  const int half = lane >> 4, l15 = lane & 15, hi8 = half * 8;
  const int row0 = blockIdx.x * QR_BLK + wave * QR_WAVE;
  const size_t bbase = (size_t)b * NTOK;

  // Q A-fragment: 16 rows x full head dim 32, loaded once.
  const v16bf aq = load_frag_a(qb + (bbase + row0 + l15) * DHEAD, hi8);

  v8f o[16];
#pragma unroll
  for (int t = 0; t < 16; ++t) o[t] = (v8f){0.f,0.f,0.f,0.f,0.f,0.f,0.f,0.f};
  float m[8], lsum[8];
#pragma unroll
  for (int r = 0; r < 8; ++r) { m[r] = -3.0e38f; lsum[r] = 0.f; }

  for (int kc = 0; kc < NTOK / KCHUNK; ++kc) {
    const size_t krow0 = bbase + (size_t)kc * KCHUNK;  // first key row this chunk
    __syncthreads();                                   // prev-iter LDS reads done

    { // K chunk -> LDS, straight copy (32x32 bf16 = 128 uint4)
      const u32x4* src = (const u32x4*)(kb + krow0 * DHEAD);
      ((u32x4*)Klds)[tid] = src[tid];
    }
    { // V chunk (already transposed in global): 256 rows x 32 keys = 1024 uint4
#pragma unroll
      for (int it = 0; it < (CDIM * KCHUNK * 2 / 16) / 128; ++it) {  // 8 iters
        int i = it * 128 + tid;             // i = c*4 + q
        int c = i >> 2, q = i & 3;
        ((u32x4*)Vt)[i] = *(const u32x4*)(vT + (size_t)c * MTOT + krow0 + q * 8);
      }
    }
    // Prefetch next chunk while this one is consumed (global_prefetch_b8).
    if (kc + 1 < NTOK / KCHUNK) {
      const size_t nrow0 = krow0 + KCHUNK;
      __builtin_prefetch(vT + (size_t)(2 * tid) * MTOT + nrow0, 0, 1);
      __builtin_prefetch(vT + (size_t)(2 * tid + 1) * MTOT + nrow0, 0, 1);
      __builtin_prefetch(kb + nrow0 * DHEAD + tid * 8, 0, 1);
    }
    __syncthreads();

    // S = q . k^T for 32 keys: two 16x16 WMMA tiles (K = DHEAD = 32).
    v8f s0 = {0.f,0.f,0.f,0.f,0.f,0.f,0.f,0.f};
    v8f s1 = {0.f,0.f,0.f,0.f,0.f,0.f,0.f,0.f};
    {
      v16bf bk0 = load_frag32(Klds + l15 * DHEAD + half * 16);        // keys 0-15
      v16bf bk1 = load_frag32(Klds + (16 + l15) * DHEAD + half * 16); // keys 16-31
      s0 = __builtin_amdgcn_wmma_f32_16x16x32_bf16(
          false, aq, false, bk0, (short)0, s0, false, false);
      s1 = __builtin_amdgcn_wmma_f32_16x16x32_bf16(
          false, aq, false, bk1, (short)0, s1, false, false);
    }

    // Online softmax. Row r+8*half lives in one 16-lane half of the wave;
    // xor-shuffles with mask<=8 reduce within that half.
    float p0[8], p1[8];
#pragma unroll
    for (int r = 0; r < 8; ++r) {
      float mx = fmaxf(s0[r], s1[r]);
#pragma unroll
      for (int off = 8; off >= 1; off >>= 1) mx = fmaxf(mx, __shfl_xor(mx, off, 32));
      float nm = fmaxf(m[r], mx);
      float alpha = __expf(m[r] - nm);
      float e0 = __expf(s0[r] - nm);
      float e1 = __expf(s1[r] - nm);
      float rs = e0 + e1;
#pragma unroll
      for (int off = 8; off >= 1; off >>= 1) rs += __shfl_xor(rs, off, 32);
      lsum[r] = lsum[r] * alpha + rs;
      m[r] = nm;
      p0[r] = e0; p1[r] = e1;
#pragma unroll
      for (int t = 0; t < 16; ++t) o[t][r] *= alpha;
    }

    // P (C-layout) -> per-wave LDS scratch -> A-fragment layout.
    unsigned short* pw = Plds[wave];
#pragma unroll
    for (int r = 0; r < 8; ++r) {
      int row = r + hi8;
      pw[row * KCHUNK + l15] = f2bf(p0[r]);
      pw[row * KCHUNK + 16 + l15] = f2bf(p1[r]);
    }
    __syncthreads();

    const v16bf ap = load_frag_a(pw + l15 * KCHUNK, hi8);

    // O += P . V : 16 output tiles across C=256.
#pragma unroll
    for (int t = 0; t < 16; ++t) {
      v16bf bv = load_frag32(Vt + (t * 16 + l15) * KCHUNK + half * 16);
      o[t] = __builtin_amdgcn_wmma_f32_16x16x32_bf16(
          false, ap, false, bv, (short)0, o[t], false, false);
    }
  }

  // Epilogue: out = x + gamma * (O / l)
  const float g = gamma[0];
#pragma unroll
  for (int t = 0; t < 16; ++t) {
#pragma unroll
    for (int r = 0; r < 8; ++r) {
      int row = row0 + r + hi8;
      int ch = t * 16 + l15;
      size_t idx = (bbase + row) * CDIM + ch;
      out[idx] = x[idx] + g * (o[t][r] / lsum[r]);
    }
  }
}

// ---------------------------------------------------------------- launch
extern "C" void kernel_launch(void* const* d_in, const int* in_sizes, int n_in,
                              void* d_out, int out_size, void* d_ws, size_t ws_size,
                              hipStream_t stream) {
  (void)in_sizes; (void)n_in; (void)out_size; (void)ws_size;
  const float* x  = (const float*)d_in[0];
  const float* Wq = (const float*)d_in[1];
  const float* Wk = (const float*)d_in[2];
  const float* Wv = (const float*)d_in[3];
  const float* gm = (const float*)d_in[4];
  float* out = (float*)d_out;

  char* ws = (char*)d_ws;
  size_t off = 0;
  unsigned short* xb  = (unsigned short*)(ws + off); off += (size_t)MTOT * CDIM * 2;
  unsigned short* wqt = (unsigned short*)(ws + off); off += (size_t)CDIM * DHEAD * 2;
  unsigned short* wkt = (unsigned short*)(ws + off); off += (size_t)CDIM * DHEAD * 2;
  unsigned short* wvt = (unsigned short*)(ws + off); off += (size_t)CDIM * CDIM * 2;
  unsigned short* qb  = (unsigned short*)(ws + off); off += (size_t)MTOT * DHEAD * 2;
  unsigned short* kbp = (unsigned short*)(ws + off); off += (size_t)MTOT * DHEAD * 2;
  unsigned short* vTp = (unsigned short*)(ws + off); off += (size_t)MTOT * CDIM * 2;

  const int npairs = MTOT * CDIM / 2;
  cvt_f32_bf16x2<<<(npairs + 255) / 256, 256, 0, stream>>>(x, (unsigned int*)xb, npairs);
  cvt_w_transpose<<<(CDIM * DHEAD + 255) / 256, 256, 0, stream>>>(Wq, wqt, CDIM, DHEAD);
  cvt_w_transpose<<<(CDIM * DHEAD + 255) / 256, 256, 0, stream>>>(Wk, wkt, CDIM, DHEAD);
  cvt_w_transpose<<<(CDIM * CDIM + 255) / 256, 256, 0, stream>>>(Wv, wvt, CDIM, CDIM);

  const int tiles_m = MTOT / 16;   // 2048
  gemm_bf16<<<tiles_m * 2 / 4,  128, 0, stream>>>(xb, wqt, qb,  DHEAD, 2, 0);
  gemm_bf16<<<tiles_m * 2 / 4,  128, 0, stream>>>(xb, wkt, kbp, DHEAD, 2, 0);
  gemm_bf16<<<tiles_m * 16 / 4, 128, 0, stream>>>(xb, wvt, vTp, CDIM, 16, 1);

  dim3 grid(NTOK / QR_BLK, BATCH);
  flash_attn<<<grid, 128, 0, stream>>>(qb, kbp, vTp, x, gm, out);
}
